// Ensemble_KF_Unimodal_Fusion_6837587936053
// MI455X (gfx1250) — matrix operand order repaired and checked by reading, not compiled, not run on hardware
//
#include <hip/hip_runtime.h>
#include <hip/hip_bf16.h>
#include <stdint.h>

// ---------------------------------------------------------------------------
// CDNA5 (gfx1250) Ensemble-KF forward.
//  - All large MLP GEMMs via V_WMMA_F32_16X16X32_BF16 (wave32 WMMA), BK=64.
//  - Split-K (x16) for the wide img_enc layer-1 GEMM to fill the chip.
//  - Stage-C per-batch tiles loaded via Tensor Data Mover (tensor_load_to_lds)
//    with D# LDS padding, synced with s_wait_tensorcnt.
// ---------------------------------------------------------------------------

typedef __attribute__((ext_vector_type(16))) __bf16 v16bf;
typedef __attribute__((ext_vector_type(8)))  __bf16 bf16x8;
typedef __attribute__((ext_vector_type(4)))  __bf16 bf16x4;
typedef __attribute__((ext_vector_type(8)))  float  v8f;
typedef __attribute__((ext_vector_type(4)))  unsigned int u32x4;
typedef __attribute__((ext_vector_type(8)))  int i32x8;
typedef __attribute__((ext_vector_type(4)))  int i32x4;

__device__ __forceinline__ __bf16 f2bf(float f) {
  unsigned u = __builtin_bit_cast(unsigned, f);
  unsigned short h = (unsigned short)(u >> 16); // truncate to bf16
  return __builtin_bit_cast(__bf16, h);
}

union FragBF { bf16x8 h[2]; v16bf v; };

#define BM 128
#define BN 64
#define BK 64
#define LDP 72  // LDS row pitch in bf16: 144B (16B-aligned, bank-skewed)

// ---------------------------------------------------------------------------
// Generic GEMM: C[M,N] = act(A[M,K] @ W[K,N] + bias) [+ res], bf16 WMMA.
// Requires M % 128 == 0, K % 32 == 0 (K tail of 32 zero-padded). N guarded.
// If `partial` != null: split-K mode. Block z covers K range
// [z*kchunk, min(K,(z+1)*kchunk)) and stores raw f32 partials to
// partial + z*M*N (no bias/relu/res).
// ---------------------------------------------------------------------------
__global__ __launch_bounds__(256)
void ekf_gemm_bf16(const float* __restrict__ A, const float* __restrict__ W,
                   const float* __restrict__ bias, const float* __restrict__ res,
                   float* __restrict__ C, int M, int N, int K, int relu,
                   int kchunk, float* __restrict__ partial)
{
  __shared__ __attribute__((aligned(16))) __bf16 As[BM][LDP];
  __shared__ __attribute__((aligned(16))) __bf16 Ws[BN][LDP];

  const int t    = threadIdx.x;
  const int lane = t & 31;
  const int wave = t >> 5;
  const int m0   = blockIdx.x * BM;
  const int n0   = blockIdx.y * BN;

  int kbeg = 0, kend = K;
  if (partial) {
    kbeg = blockIdx.z * kchunk;
    kend = (kbeg + kchunk < K) ? (kbeg + kchunk) : K;
  }

  v8f acc[4];
  acc[0] = {}; acc[1] = {}; acc[2] = {}; acc[3] = {};

  const int wn  = t & 63;                 // W loader: column within tile
  const int wkb = (t >> 6) << 4;          // W loader: k-subblock base (0/16/32/48)
  const int ncl = (n0 + wn < N) ? (n0 + wn) : (N - 1);   // clamped, branchless
  const bool nok = (n0 + wn) < N;

  for (int k0 = kbeg; k0 < kend; k0 += BK) {
    // ---- stage A tile: 128x64 f32, 8 float4 per thread, loads batched ----
    float4 va[8];
#pragma unroll
    for (int i = 0; i < 8; i++) {
      int idx = t + i * 256;               // 0..2047 float4 slots
      int r   = idx >> 4;                  // row 0..127 (16 float4 per row)
      int c4  = (idx & 15) << 2;           // col 0..60 step 4
      int kk  = k0 + c4;
      const float* ap = A + (size_t)(m0 + r) * K + (kk < kend ? kk : kbeg);
      float4 v = *(const float4*)ap;
      if (kk >= kend) { v.x = 0.f; v.y = 0.f; v.z = 0.f; v.w = 0.f; }
      va[i] = v;
    }
#pragma unroll
    for (int i = 0; i < 8; i++) {
      int idx = t + i * 256;
      int r   = idx >> 4;
      int c4  = (idx & 15) << 2;
      union { bf16x4 v; __bf16 e[4]; } p;
      p.e[0] = f2bf(va[i].x); p.e[1] = f2bf(va[i].y);
      p.e[2] = f2bf(va[i].z); p.e[3] = f2bf(va[i].w);
      *(bf16x4*)&As[r][c4] = p.v;
    }
    // ---- stage W tile transposed: Ws[n][k], packed b128 stores ----
    {
      float wv[16];
#pragma unroll
      for (int j = 0; j < 16; j++) {
        int k   = k0 + wkb + j;
        int krd = (k < kend) ? k : (kend - 1);
        float w = W[(size_t)krd * N + ncl];
        wv[j] = (nok && k < kend) ? w : 0.0f;   // branchless select
      }
      union { bf16x8 v; __bf16 e[8]; } p0, p1;
#pragma unroll
      for (int j = 0; j < 8; j++) { p0.e[j] = f2bf(wv[j]); p1.e[j] = f2bf(wv[j + 8]); }
      *(bf16x8*)&Ws[wn][wkb]     = p0.v;
      *(bf16x8*)&Ws[wn][wkb + 8] = p1.v;
    }
    __syncthreads();

    // gfx1250 prefetch of next K-tile (global_prefetch_b8)
    if (k0 + BK < kend) {
      if (t < 128) {
        __builtin_prefetch(A + (size_t)(m0 + t) * K + k0 + BK, 0, 0);
      } else if (t < 192) {
        int rr = t - 128;
        __builtin_prefetch(W + (size_t)(k0 + BK + rr) * N + n0, 0, 0);
      }
    }

    // ---- compute: 2 x (A-frag + 4 WMMA) ----
#pragma unroll
    for (int kk = 0; kk < BK; kk += 32) {
      FragBF a;
      int ar = wave * 16 + (lane & 15);
      int ak = kk + ((lane >> 4) << 3);            // +0 or +8
      a.h[0] = *(const bf16x8*)&As[ar][ak];        // K 0..7 / 8..15
      a.h[1] = *(const bf16x8*)&As[ar][ak + 16];   // K 16..23 / 24..31
#pragma unroll
      for (int j = 0; j < 4; j++) {
        FragBF b;
        int bn = (j << 4) + (lane & 15);
        int bk = kk + ((lane >> 4) << 3);          // +0 or +8
        b.h[0] = *(const bf16x8*)&Ws[bn][bk];
        b.h[1] = *(const bf16x8*)&Ws[bn][bk + 16];
        acc[j] = __builtin_amdgcn_wmma_f32_16x16x32_bf16(
            false, a.v, false, b.v, (short)0, acc[j], false, false);
      }
    }
    __syncthreads();
  }

  if (partial) {
    // ---- split-K: raw partial store ----
    float* P = partial + (size_t)blockIdx.z * M * N;
#pragma unroll
    for (int j = 0; j < 4; j++) {
      int col = n0 + (j << 4) + (lane & 15);
      if (col < N) {
#pragma unroll
        for (int v = 0; v < 8; v++) {
          int row = m0 + wave * 16 + v + ((lane >> 4) << 3);
          P[(size_t)row * N + col] = acc[j][v];
        }
      }
    }
    return;
  }

  // ---- epilogue: bias / relu / residual, f32 store (lane-coalesced) ----
#pragma unroll
  for (int j = 0; j < 4; j++) {
    int col = n0 + (j << 4) + (lane & 15);
    if (col < N) {
      float bv = bias ? bias[col] : 0.0f;
#pragma unroll
      for (int v = 0; v < 8; v++) {
        int row = m0 + wave * 16 + v + ((lane >> 4) << 3);
        float o = acc[j][v] + bv;
        if (relu) o = fmaxf(o, 0.0f);
        if (res)  o += res[(size_t)row * N + col];
        C[(size_t)row * N + col] = o;
      }
    }
  }
}

// Split-K reduction: C[i] = act(sum_z partial[z*MN + i] + bias[i % N]).
__global__ __launch_bounds__(256)
void ekf_reduce_splitk(const float* __restrict__ partial, const float* __restrict__ bias,
                       float* __restrict__ C, int MN, int N, int S, int relu)
{
  int i = blockIdx.x * 256 + threadIdx.x;
  if (i < MN) {
    float s = bias[i % N];
    for (int z = 0; z < S; z++) s += partial[(size_t)z * MN + i];
    if (relu) s = fmaxf(s, 0.0f);
    C[i] = s;
  }
}

// ---------------------------------------------------------------------------
// Tensor Data Mover: 1-D f32 block -> LDS with optional per-interval padding.
// D# per CDNA5 ISA (group0: count/lds/global/type,
// group1: data_size + pad + dims/strides). 6-arg clang-23 builtin form.
// ---------------------------------------------------------------------------
#if defined(__HIP_DEVICE_COMPILE__) && __has_builtin(__builtin_amdgcn_tensor_load_to_lds)
#define EKF_TDM 1
#endif

#if defined(EKF_TDM)
__device__ __forceinline__ void tdm_load_lds(unsigned lds_addr, const void* gptr,
                                             unsigned n_elems, int pad_en,
                                             unsigned pad_interval_code)
{
  unsigned long long ga = (unsigned long long)(uintptr_t)gptr;
  union { unsigned long long q[2]; u32x4 v; } g0;
  g0.q[0] = 1ull | ((unsigned long long)lds_addr << 32);         // count=1, lds_addr
  g0.q[1] = (ga & 0x01FFFFFFFFFFFFFFull) | (2ull << 62);         // global_addr, type=2
  union { unsigned long long q[4]; i32x8 v; } g1;
  unsigned long long flags = (2ull << 16);                       // data_size = 4B
  if (pad_en)
    flags |= (1ull << 20) | ((unsigned long long)pad_interval_code << 22); // pad 1 dword
  g1.q[0] = flags | ((unsigned long long)(n_elems & 0xFFFF) << 48);   // tensor_dim0 lo
  g1.q[1] = ((unsigned long long)(n_elems >> 16) & 0xFFFF)            // tensor_dim0 hi
          | (1ull << 16)                                              // tensor_dim1 = 1
          | ((unsigned long long)(n_elems & 0xFFFF) << 48);           // tile_dim0
  g1.q[2] = 1ull                                                      // tile_dim1 = 1
          | ((unsigned long long)n_elems << 32);                      // dim0_stride lo
  g1.q[3] = 0;
  i32x4 z4 = {0, 0, 0, 0};
  i32x8 z8 = {0, 0, 0, 0, 0, 0, 0, 0};
  __builtin_amdgcn_tensor_load_to_lds(g0.v, g1.v, z4, z4, z8, 0);
}
#endif

// ---------------------------------------------------------------------------
// Stage C: per-batch Kalman update. One workgroup (256 thr) per batch element.
// ---------------------------------------------------------------------------

#define O0 0        // state_new      (B,64,32)
#define O1 524288   // m_state_new    (B,1,32)
#define O2 532480   // m_state_pred   (B,1,32)
#define O3 540672   // m1             (B,1,16)
#define O4 544768   // m2             (B,1,16)
#define O5 548864   // z              (B,1,16)
#define O6 552960   // ensemble_z     (B,64,16)
#define O7 815104   // H_X_mean       (B,1,16)

__device__ void inv16(float (*Ain)[17], float (*Out)[17], float* fac, int tid) {
  int r = tid >> 4, c = tid & 15;
  Out[r][c] = (r == c) ? 1.0f : 0.0f;
  __syncthreads();
  for (int k = 0; k < 16; k++) {
    if (tid < 16) fac[tid] = Ain[tid][k];   // snapshot column k
    __syncthreads();
    float ip = 1.0f / fac[k];
    if (r == k) { Ain[k][c] *= ip; Out[k][c] *= ip; }
    __syncthreads();
    if (r != k) {
      float f = fac[r] * ip;
      Ain[r][c] -= f * Ain[k][c];
      Out[r][c] -= f * Out[k][c];
    }
    __syncthreads();
  }
}

__global__ __launch_bounds__(256)
void ekf_stage_c(const float* __restrict__ SPg, const float* __restrict__ HXg,
                 const float* __restrict__ E1g, const float* __restrict__ E2g,
                 const float* __restrict__ IEg,
                 const float* __restrict__ Wn1, const float* __restrict__ bn1,
                 const float* __restrict__ Wn2, const float* __restrict__ bn2,
                 float* __restrict__ out)
{
  __shared__ float sp[64][33];
  __shared__ float hx[64][17];
  __shared__ float ha[64][17];
  __shared__ float e1[64][17];
  __shared__ float e2[64][17];
  __shared__ float ie[64][17];
  __shared__ float ez[64][17];
  __shared__ float c1[16][17], c2[16][17];
  __shared__ float ic1[16][17], ic2[16][17], cf[16][17];
  __shared__ float inn[16][17], iinn[16][17];
  __shared__ float ah[32][17], km[32][17];
  __shared__ float mA[32], mH[16], m1v[16], m2v[16], zv[16], dg[16], hb[32], fac[16];

  const int tid = threadIdx.x;
  const int b   = blockIdx.x;

  // ---- load per-batch tiles: TDM DMA with D# padding, else VALU loads ----
#if defined(EKF_TDM)
  if (tid == 0) {
    // pad_interval codes: 3 -> every 16 dwords, 4 -> every 32 dwords (+1 dword)
    tdm_load_lds((unsigned)(uintptr_t)(void*)&sp[0][0], SPg + (size_t)b * 2048, 2048, 1, 4);
    tdm_load_lds((unsigned)(uintptr_t)(void*)&hx[0][0], HXg + (size_t)b * 1024, 1024, 1, 3);
    tdm_load_lds((unsigned)(uintptr_t)(void*)&e1[0][0], E1g + (size_t)b * 1024, 1024, 1, 3);
    tdm_load_lds((unsigned)(uintptr_t)(void*)&e2[0][0], E2g + (size_t)b * 1024, 1024, 1, 3);
    tdm_load_lds((unsigned)(uintptr_t)(void*)&ie[0][0], IEg,                    1024, 1, 3);
#if __has_builtin(__builtin_amdgcn_s_wait_tensorcnt)
    __builtin_amdgcn_s_wait_tensorcnt(0);
#endif
  }
  __syncthreads();
#else
  for (int i = tid; i < 2048; i += 256) sp[i >> 5][i & 31] = SPg[(size_t)b * 2048 + i];
  for (int i = tid; i < 1024; i += 256) {
    int n = i >> 4, j = i & 15;
    hx[n][j] = HXg[(size_t)b * 1024 + i];
    e1[n][j] = E1g[(size_t)b * 1024 + i];
    e2[n][j] = E2g[(size_t)b * 1024 + i];
    ie[n][j] = IEg[i];
  }
  __syncthreads();
#endif

  // ---- means ----
  if (tid < 32) {
    float s = 0; for (int n = 0; n < 64; n++) s += sp[n][tid];
    mA[tid] = s * (1.0f / 64.0f);
  } else if (tid < 48) {
    int j = tid - 32; float s = 0; for (int n = 0; n < 64; n++) s += hx[n][j];
    mH[j] = s * (1.0f / 64.0f);
  } else if (tid < 64) {
    int j = tid - 48; float s = 0; for (int n = 0; n < 64; n++) s += e1[n][j];
    m1v[j] = s * (1.0f / 64.0f);
  } else if (tid < 80) {
    int j = tid - 64; float s = 0; for (int n = 0; n < 64; n++) s += e2[n][j];
    m2v[j] = s * (1.0f / 64.0f);
  }
  __syncthreads();

  if (tid < 32)                out[O2 + (size_t)b * 32 + tid]        = mA[tid];
  else if (tid < 48)           out[O7 + (size_t)b * 16 + (tid - 32)] = mH[tid - 32];
  else if (tid < 64)           out[O3 + (size_t)b * 16 + (tid - 48)] = m1v[tid - 48];
  else if (tid < 80)           out[O4 + (size_t)b * 16 + (tid - 64)] = m2v[tid - 64];

  // ---- centering ----
  for (int i = tid; i < 1024; i += 256) {
    int n = i >> 4, j = i & 15;
    ha[n][j] = hx[n][j] - mH[j];
    e1[n][j] -= m1v[j];
    e2[n][j] -= m2v[j];
  }
  __syncthreads();

  // ---- covariances (DZ x DZ over EN) ----
  {
    int i = tid >> 4, j = tid & 15;
    float s1 = 0, s2 = 0;
    for (int n = 0; n < 64; n++) { s1 += e1[n][i] * e1[n][j]; s2 += e2[n][i] * e2[n][j]; }
    c1[i][j] = s1 * (1.0f / 63.0f);
    c2[i][j] = s2 * (1.0f / 63.0f);
  }
  __syncthreads();

  inv16(c1, ic1, fac, tid);
  inv16(c2, ic2, fac, tid);
  { int i = tid >> 4, j = tid & 15; c1[i][j] = ic1[i][j] + ic2[i][j]; }
  __syncthreads();
  inv16(c1, cf, fac, tid);   // cov_fuse

  // ---- ensemble_z = init_e @ cov_fuse ----
  for (int i = tid; i < 1024; i += 256) {
    int n = i >> 4, j = i & 15;
    float s = 0;
    for (int k = 0; k < 16; k++) s += ie[n][k] * cf[k][j];
    ez[n][j] = s;
    out[O6 + (size_t)b * 1024 + i] = s;
  }
  __syncthreads();
  if (tid < 16) {
    float s = 0; for (int n = 0; n < 64; n++) s += ez[n][tid];
    zv[tid] = s * (1.0f / 64.0f);
    out[O5 + (size_t)b * 16 + tid] = zv[tid];
  }
  __syncthreads();

  // ---- noise MLP: 16 -> 32 (relu) -> 16 ; diag = (o+1e-3)^2 + sqrt(0.009) ----
  if (tid < 32) {
    float s = bn1[tid];
    for (int i = 0; i < 16; i++) s += zv[i] * Wn1[i * 32 + tid];
    hb[tid] = fmaxf(s, 0.0f);
  }
  __syncthreads();
  if (tid < 16) {
    float s = bn2[tid];
    for (int i = 0; i < 32; i++) s += hb[i] * Wn2[i * 16 + tid];
    s += 0.001f;
    dg[tid] = s * s + 0.09486832980505137f;
  }
  __syncthreads();

  // ---- innovation = H_A^T H_A / 63 + R ----
  {
    int i = tid >> 4, j = tid & 15;
    float s = 0;
    for (int n = 0; n < 64; n++) s += ha[n][i] * ha[n][j];
    inn[i][j] = s * (1.0f / 63.0f) + ((i == j) ? dg[i] : 0.0f);
  }
  __syncthreads();
  inv16(inn, iinn, fac, tid);

  // ---- AH = A @ H_A / 63  (32x16) ----
  for (int idx = tid; idx < 512; idx += 256) {
    int d = idx >> 4, j = idx & 15;
    float md = mA[d], s = 0;
    for (int n = 0; n < 64; n++) s += (sp[n][d] - md) * ha[n][j];
    ah[d][j] = s * (1.0f / 63.0f);
  }
  __syncthreads();

  // ---- K = AH @ inv(innovation) ----
  for (int idx = tid; idx < 512; idx += 256) {
    int d = idx >> 4, j = idx & 15;
    float s = 0;
    for (int k = 0; k < 16; k++) s += ah[d][k] * iinn[k][j];
    km[d][j] = s;
  }
  __syncthreads();

  // ---- gain + state_new ----
  for (int idx = tid; idx < 2048; idx += 256) {
    int n = idx >> 5, d = idx & 31;
    float s = 0;
    for (int j = 0; j < 16; j++) s += km[d][j] * (ez[n][j] - hx[n][j]);
    float sn = sp[n][d] + s;
    sp[n][d] = sn;
    out[O0 + (size_t)b * 2048 + idx] = sn;
  }
  __syncthreads();
  if (tid < 32) {
    float s = 0; for (int n = 0; n < 64; n++) s += sp[n][tid];
    out[O1 + (size_t)b * 32 + tid] = s * (1.0f / 64.0f);
  }
}

// ---------------------------------------------------------------------------
// Launcher
// ---------------------------------------------------------------------------
static inline void gemm(const float* A, const float* W, const float* bias,
                        const float* res, float* C, int M, int N, int K,
                        int relu, hipStream_t s)
{
  dim3 grid(M / BM, (N + BN - 1) / BN);
  ekf_gemm_bf16<<<grid, 256, 0, s>>>(A, W, bias, res, C, M, N, K, relu, 0, nullptr);
}

extern "C" void kernel_launch(void* const* d_in, const int* in_sizes, int n_in,
                              void* d_out, int out_size, void* d_ws, size_t ws_size,
                              hipStream_t stream)
{
  (void)in_sizes; (void)n_in; (void)out_size; (void)ws_size;
#define IN(i) ((const float*)d_in[i])
  const float* obs_img   = IN(0);   // (256, 3*64*64) flat
  const float* obs_1     = IN(1);   // (256, 128)
  const float* state_old = IN(2);   // (256*64, 32) flat
  // d_in[3] = m_state (unused by reference forward)
  const float* init_e    = IN(4);   // (64, 16)
  // proc
  const float *Wp1 = IN(5),  *bp1 = IN(6),  *Wp2 = IN(7),  *bp2 = IN(8);
  const float *Wp3 = IN(9),  *bp3 = IN(10), *Wp4 = IN(11), *bp4 = IN(12);
  // obs
  const float *Wo1 = IN(13), *bo1 = IN(14), *Wo2 = IN(15), *bo2 = IN(16);
  const float *Wo3 = IN(17), *bo3 = IN(18), *Wo4 = IN(19), *bo4 = IN(20);
  const float *Wo5 = IN(21), *bo5 = IN(22);
  // noise
  const float *Wn1 = IN(23), *bn1 = IN(24), *Wn2 = IN(25), *bn2 = IN(26);
  // img_enc
  const float *Wi1 = IN(27), *bi1 = IN(28), *Wi2 = IN(29), *bi2 = IN(30);
  // sens_enc
  const float *Vs1 = IN(31), *cs1 = IN(32), *Vs2 = IN(33), *cs2 = IN(34);
#undef IN

  float* ws = (float*)d_ws;
  float* S1 = ws;                  // 4,194,304 floats (16 MB)
  float* S2 = ws + 4194304;        // 8,388,608 floats (32 MB)
  float* SP = ws + 12582912;       // state_pred 16384x32
  float* HX = ws + 13107200;       // H_X 16384x16
  float* E1 = ws + 13369344;       // ens1 256x1024
  float* E2 = ws + 13631488;       // ens2 256x1024
  float* IH = ws + 13893632;       // img hidden 256x256
  float* SH = ws + 13959168;       // sens hidden 256x256
  float* PK = ws + 14024704;       // split-K partials: 16 x 65536 floats

  const int MBE = 256 * 64;        // 16384

  // proc MLP + residual -> state_pred
  gemm(state_old, Wp1, bp1, nullptr, S1, MBE, 64,  32,  1, stream);
  gemm(S1,        Wp2, bp2, nullptr, S2, MBE, 512, 64,  1, stream);
  gemm(S2,        Wp3, bp3, nullptr, S1, MBE, 256, 512, 1, stream);
  gemm(S1,        Wp4, bp4, state_old, SP, MBE, 32, 256, 0, stream);

  // obs MLP -> H_X
  gemm(SP, Wo1, bo1, nullptr, S1, MBE, 64,  32,  1, stream);
  gemm(S1, Wo2, bo2, nullptr, S2, MBE, 128, 64,  1, stream);
  gemm(S2, Wo3, bo3, nullptr, S1, MBE, 128, 128, 1, stream);
  gemm(S1, Wo4, bo4, nullptr, S2, MBE, 64,  128, 1, stream);
  gemm(S2, Wo5, bo5, nullptr, HX, MBE, 16,  64,  0, stream);

  // image encoder layer 1: split-K x16 (K=12288 -> 16 x 768) for parallelism
  {
    const int S = 16, CHUNK = 768, MN = 256 * 256;
    dim3 grid(256 / BM, 256 / BN, S);   // 2 x 4 x 16 = 128 workgroups
    ekf_gemm_bf16<<<grid, 256, 0, stream>>>(obs_img, Wi1, nullptr, nullptr,
                                            nullptr, 256, 256, 12288, 0, CHUNK, PK);
    ekf_reduce_splitk<<<(MN + 255) / 256, 256, 0, stream>>>(PK, bi1, IH, MN, 256, S, 1);
  }
  gemm(IH, Wi2, bi2, nullptr, E1, 256, 1024, 256, 0, stream);

  // sensor encoder -> ens2
  gemm(obs_1, Vs1, cs1, nullptr, SH, 256, 256,  128, 1, stream);
  gemm(SH,    Vs2, cs2, nullptr, E2, 256, 1024, 256, 0, stream);

  // per-batch Kalman update + all outputs
  ekf_stage_c<<<dim3(256), 256, 0, stream>>>(SP, HX, E1, E2, init_e,
                                             Wn1, bn1, Wn2, bn2, (float*)d_out);
}
